// FastInstDecoder_35261681500621
// MI455X (gfx1250) — compile-verified, matrix operand order checked by reading, not run on hardware
//
#include <hip/hip_runtime.h>
#include <hip/hip_bf16.h>
#include <math.h>

typedef __attribute__((ext_vector_type(16))) __bf16        v16bf;
typedef __attribute__((ext_vector_type(8)))  float         v8f;
typedef __attribute__((ext_vector_type(4)))  unsigned int  u32x4;
typedef __attribute__((ext_vector_type(4)))  int           v4i;

union ABFrag { v16bf v; u32x4 q[2]; };

#define B_    8
#define C_    256
#define H_    100
#define W_    152
#define HW_   15200
#define NPIX  (B_ * HW_)      // 121600
#define NC1   81
#define NCP   96
#define TOPK_ 100

// ---------------- workspace layout (bytes) ----------------
#define OFF_XB   ((size_t)0)           // 8*100*152*256 bf16 = 62,259,200 B
#define OFF_HIDB ((size_t)62259200)    // same size
#define OFF_WB1  ((size_t)124518400)   // 9*256*256 bf16   = 1,179,648 B
#define OFF_WB2  ((size_t)125698048)   // 96*256 bf16      = 49,152 B
#define OFF_P    ((size_t)125747200)   // 121600 f32
#define OFF_I    ((size_t)126233600)   // 121600 i32
#define OFF_S    ((size_t)126720000)   // 121600 f32
#define OFF_T    ((size_t)127206400)   // 8*100 i32

// ---------- CDNA5 async global->LDS copy (ASYNCcnt path) ----------
#if defined(__has_builtin)
#if __has_builtin(__builtin_amdgcn_global_load_async_to_lds_b128)
#define HAVE_ASYNC_LDS 1
#endif
#endif

typedef __attribute__((address_space(1))) v4i* gptr_v4i;
typedef __attribute__((address_space(3))) v4i* lptr_v4i;

__device__ __forceinline__ void cp16_async(const unsigned short* g, unsigned short* l) {
#ifdef HAVE_ASYNC_LDS
  __builtin_amdgcn_global_load_async_to_lds_b128((gptr_v4i)g, (lptr_v4i)l, 0, 0);
#else
  *(u32x4*)l = *(const u32x4*)g;
#endif
}
__device__ __forceinline__ void async_wait() {
#ifdef HAVE_ASYNC_LDS
  asm volatile("s_wait_asynccnt 0x0" ::: "memory");
#endif
}

__device__ __forceinline__ unsigned short f2bf(float f) {
  unsigned int u = __float_as_uint(f);
  unsigned int lsb = (u >> 16) & 1u;
  u += 0x7fffu + lsb;               // round-to-nearest-even
  return (unsigned short)(u >> 16);
}

// ---- x: NCHW f32 -> NHWC bf16 (tiled transpose) ----
__global__ void k_xpose(const float* __restrict__ x, unsigned short* __restrict__ xb) {
  __shared__ float t[16][17];
  int ny = blockIdx.z;                    // n*H_ + y
  int n = ny / H_, y = ny % H_;
  int xr = blockIdx.x * 16 + threadIdx.x;
  int cr = blockIdx.y * 16 + threadIdx.y;
  t[threadIdx.y][threadIdx.x] = (xr < W_) ? x[((size_t)(n * C_ + cr) * H_ + y) * W_ + xr] : 0.f;
  __syncthreads();
  int xw = blockIdx.x * 16 + threadIdx.y;
  int cw = blockIdx.y * 16 + threadIdx.x;
  if (xw < W_)
    xb[((size_t)(n * H_ + y) * W_ + xw) * C_ + cw] = f2bf(t[threadIdx.x][threadIdx.y]);
}

// ---- repack weights to bf16: wb1[tap][oc][ic], wb2[cls(96)][ic] ----
__global__ void k_wpack(const float* __restrict__ w1, const float* __restrict__ w2,
                        unsigned short* __restrict__ wb1, unsigned short* __restrict__ wb2) {
  int t = blockIdx.x * blockDim.x + threadIdx.x;
  const int NW1 = 9 * C_ * C_;
  if (t < NW1) {
    int kk = t / (C_ * C_);
    int oc = (t / C_) % C_;
    int ic = t % C_;
    wb1[(size_t)(kk * C_ + oc) * C_ + ic] = f2bf(w1[(size_t)(oc * C_ + ic) * 9 + kk]);
  }
  if (t < NCP * C_) {
    int cls = t / C_, ic = t % C_;
    wb2[t] = f2bf(cls < NC1 ? w2[cls * C_ + ic] : 0.f);
  }
}

// stage one B tile row (oc=tid) for flat K-step s: [256 oc][32 ic] -> lds row (stride 40)
__device__ __forceinline__ void stage_b_conv3(const unsigned short* __restrict__ wb1,
                                              unsigned short* dst_row, int s, int tid) {
  int kk  = (s % 18) >> 1;
  int icc = (s / 18) * 2 + (s & 1);
  const unsigned short* src = &wb1[(size_t)(kk * C_ + tid) * C_ + icc * 32];
  cp16_async(src,      dst_row);
  cp16_async(src + 8,  dst_row + 8);
  cp16_async(src + 16, dst_row + 16);
  cp16_async(src + 24, dst_row + 24);
}

// ---- 3x3 conv as implicit GEMM, bf16 WMMA, f32 accum ----
// grid (3 x-tiles, 100 y, 8 n), block 256 (8 waves)
// flat K-step s = ic4*18 + kk*2 + ics  (ic4: 64-ch input slice, kk: tap, ics: 32-ch half)
__global__ __launch_bounds__(256) void k_conv3(
    const unsigned short* __restrict__ xb, const unsigned short* __restrict__ wb1,
    const float* __restrict__ b1, unsigned short* __restrict__ hidb) {
  __shared__ alignas(16) unsigned short smH[64 * 264];   // halo slice (rows stride 72) / epilogue (stride 264)
  __shared__ alignas(16) unsigned short lb[2][256 * 40]; // double-buffered B, padded rows
  const int tid = threadIdx.x;
  const int x0 = blockIdx.x * 64, y = blockIdx.y, n = blockIdx.z;
  const int lane = tid & 31, wv = tid >> 5;
  const int msub = wv & 3, nh = wv >> 2;
  const int lr = lane & 15, lhf = lane >> 4;

  v8f acc[8];
  for (int t = 0; t < 8; ++t) for (int i = 0; i < 8; ++i) acc[t][i] = 0.f;

  stage_b_conv3(wb1, &lb[0][tid * 40], 0, tid);   // prologue prefetch (waited at first barrier)

  for (int s = 0; s < 72; ++s) {
    const int buf = s & 1;
    if ((s % 18) == 0) {
      // stage 3x66 halo of ic slice (128B/pixel = 8 chunks, 32 pixels in flight)
      const int ic4 = s / 18;
      const int c4 = tid & 7;
      for (int p = tid >> 3; p < 198; p += 32) {
        int row = p / 66, col = p % 66;
        int yy = y + row - 1, xx = x0 + col - 1;
        unsigned short* dst = &smH[(row * 66 + col) * 72 + c4 * 8];
        if (yy >= 0 && yy < H_ && xx >= 0 && xx < W_) {
          cp16_async(&xb[((size_t)(n * H_ + yy) * W_ + xx) * C_ + ic4 * 64 + c4 * 8], dst);
        } else {
          u32x4 z = {0u, 0u, 0u, 0u};
          *(u32x4*)dst = z;
        }
      }
      async_wait();
      __syncthreads();
    }
    if (s + 1 < 72) stage_b_conv3(wb1, &lb[buf ^ 1][tid * 40], s + 1, tid);  // overlap with WMMA

    const int kk = (s % 18) >> 1, ky = kk / 3, kx = kk % 3, ics = s & 1;
    ABFrag a;
    {
      const int ml = msub * 16 + lr;
      const unsigned short* ap = &smH[(ky * 66 + ml + kx) * 72 + ics * 32 + lhf * 8];
      a.q[0] = *(const u32x4*)ap;
      a.q[1] = *(const u32x4*)(ap + 16);
    }
    // issue ALL 16 B-fragment ds_loads first (all 8 frags live in regs), then
    // the 8 WMMAs: in-order DScnt lets the compiler use partial waits so the
    // loads stream underneath the retiring WMMAs
    const unsigned short* brow = &lb[buf][(nh * 128 + lr) * 40 + lhf * 8];
    ABFrag bfr[8];
#pragma unroll
    for (int t = 0; t < 8; ++t) {
      const unsigned short* bp = brow + t * 16 * 40;
      bfr[t].q[0] = *(const u32x4*)bp;
      bfr[t].q[1] = *(const u32x4*)(bp + 16);
    }
#pragma unroll
    for (int t = 0; t < 8; ++t)
      acc[t] = __builtin_amdgcn_wmma_f32_16x16x32_bf16(
          false, a.v, false, bfr[t].v, (short)0, acc[t], false, false);

    async_wait();
    __syncthreads();
  }

  // epilogue: bias + relu -> padded LDS tile [64 px][256 oc] (stride 264), then b128 stores
  for (int t = 0; t < 8; ++t) {
    int oc = nh * 128 + t * 16 + lr;
    float bias = b1[oc];
    for (int r = 0; r < 8; ++r) {
      int px = msub * 16 + r + lhf * 8;
      float v = acc[t][r] + bias;
      smH[px * 264 + oc] = f2bf(v > 0.f ? v : 0.f);
    }
  }
  __syncthreads();
  {
    int px = tid & 63, q = tid >> 6;
    int gx = x0 + px;
    if (gx < W_) {
      u32x4* dst = (u32x4*)&hidb[((size_t)(n * H_ + y) * W_ + gx) * C_ + q * 64];
      for (int i = 0; i < 8; ++i)
        dst[i] = *(const u32x4*)&smH[px * 264 + q * 64 + i * 8];
    }
  }
}

// ---- 1x1 conv (256 -> 81, N padded 96) + bias -> logits NCHW f32 ----
// grid 950, block 256; 128 pixels/WG, wave = one 16-pixel M-subtile x 6 N-tiles
__global__ __launch_bounds__(256) void k_conv2(
    const unsigned short* __restrict__ hidb, const unsigned short* __restrict__ wb2,
    const float* __restrict__ b2, float* __restrict__ logits) {
  __shared__ alignas(16) unsigned short lbw[NCP * 264];  // 50,688 B: B (padded) then f32 [96][132] epilogue tile
  const int tid = threadIdx.x;
  for (int i = tid; i < NCP * 32; i += 256) {            // 96 rows * 32 chunks of 16B
    int row = i >> 5, ch = i & 31;
    cp16_async(&wb2[row * C_ + ch * 8], &lbw[row * 264 + ch * 8]);
  }
  async_wait();
  __syncthreads();

  const int lane = tid & 31, wv = tid >> 5;
  const int lr = lane & 15, lhf = lane >> 4;
  const int p0 = blockIdx.x * 128 + wv * 16;
  v8f acc[6];
  for (int t = 0; t < 6; ++t) for (int i = 0; i < 8; ++i) acc[t][i] = 0.f;

  const unsigned short* arow = &hidb[(size_t)(p0 + lr) * C_ + lhf * 8];
  ABFrag a0, a1;
  a0.q[0] = *(const u32x4*)arow;
  a0.q[1] = *(const u32x4*)(arow + 16);
#pragma unroll
  for (int icc = 0; icc < 8; ++icc) {
    ABFrag& acur = (icc & 1) ? a1 : a0;
    ABFrag& anxt = (icc & 1) ? a0 : a1;
    if (icc + 1 < 8) {  // prefetch next A fragment (global) under this step's WMMAs
      const unsigned short* ap = arow + (icc + 1) * 32;
      anxt.q[0] = *(const u32x4*)ap;
      anxt.q[1] = *(const u32x4*)(ap + 16);
    }
    // all 6 B fragments live at once -> partial DScnt waits interleave with WMMAs
    const unsigned short* brow = &lbw[lr * 264 + icc * 32 + lhf * 8];
    ABFrag bfr[6];
#pragma unroll
    for (int t = 0; t < 6; ++t) {
      const unsigned short* bp = brow + t * 16 * 264;
      bfr[t].q[0] = *(const u32x4*)bp;
      bfr[t].q[1] = *(const u32x4*)(bp + 16);
    }
#pragma unroll
    for (int t = 0; t < 6; ++t)
      acc[t] = __builtin_amdgcn_wmma_f32_16x16x32_bf16(
          false, acur.v, false, bfr[t].v, (short)0, acc[t], false, false);
  }
  __syncthreads();
  // transpose through LDS (reuse lbw as f32 [96][132]) for coalesced NCHW stores
  float* tl = (float*)lbw;
  for (int t = 0; t < 6; ++t) {
    int cls = t * 16 + lr;
    float bias = (cls < NC1) ? b2[cls] : 0.f;
    for (int r = 0; r < 8; ++r) {
      int pxl = wv * 16 + r + lhf * 8;     // 0..127
      tl[cls * 132 + pxl] = acc[t][r] + bias;
    }
  }
  __syncthreads();
  for (int i = tid; i < NC1 * 128; i += 256) {
    int cls = i >> 7, px = i & 127;
    int p = blockIdx.x * 128 + px;
    int nb = p / HW_, rem = p % HW_;
    logits[((size_t)nb * NC1 + cls) * HW_ + rem] = tl[cls * 132 + px];
  }
}

// ---- per-pixel softmax top prob + argmax over first 80 classes ----
__global__ void k_probs(const float* __restrict__ logits,
                        float* __restrict__ pmap, int* __restrict__ imap) {
  int t = blockIdx.x * blockDim.x + threadIdx.x;
  if (t >= NPIX) return;
  int n = t / HW_, rem = t % HW_;
  const float* base = logits + (size_t)n * NC1 * HW_ + rem;
  float M = -INFINITY, m80 = -INFINITY; int idx = 0;
  for (int c = 0; c < NC1; ++c) {
    float l = base[(size_t)c * HW_];
    if (l > M) M = l;
    if (c < 80 && l > m80) { m80 = l; idx = c; }
  }
  float s = 0.f;
  for (int c = 0; c < NC1; ++c) s += expf(base[(size_t)c * HW_] - M);
  pmap[t] = expf(m80 - M) / s;
  imap[t] = idx;
}

// ---- score = p + 1[local max among same-class neighbors, p>=eps] ----
__global__ void k_score(const float* __restrict__ pmap, const int* __restrict__ imap,
                        float* __restrict__ smap) {
  int t = blockIdx.x * blockDim.x + threadIdx.x;
  if (t >= NPIX) return;
  int n = t / HW_, rem = t % HW_;
  int y = rem / W_, x = rem % W_;
  float p = pmap[t]; int idx = imap[t];
  bool ismax = (p >= 1e-6f);
  if (ismax) {
    for (int dy = -1; dy <= 1; ++dy) {
      for (int dx = -1; dx <= 1; ++dx) {
        if (dy == 0 && dx == 0) continue;
        int yy = y + dy, xx = x + dx;
        if (yy < 0 || yy >= H_ || xx < 0 || xx >= W_) continue;
        int u = n * HW_ + yy * W_ + xx;
        if (imap[u] == idx && pmap[u] > p) ismax = false;
      }
    }
  }
  smap[t] = p + (ismax ? 1.f : 0.f);
}

// ---- per-batch iterative top-100 (exact top_k tie semantics) ----
__global__ __launch_bounds__(1024) void k_topk(float* __restrict__ smap, int* __restrict__ tidxv) {
  __shared__ float rv[1024];
  __shared__ int   ri[1024];
  int n = blockIdx.x, tid = threadIdx.x;
  float* sv = smap + (size_t)n * HW_;
  for (int it = 0; it < TOPK_; ++it) {
    float bv = -INFINITY; int bi = 0x7fffffff;
    for (int i = tid; i < HW_; i += 1024) {
      float v = sv[i];
      if (v > bv) { bv = v; bi = i; }   // ascending i -> first max kept
    }
    rv[tid] = bv; ri[tid] = bi;
    __syncthreads();
    for (int s = 512; s > 0; s >>= 1) {
      if (tid < s) {
        float ov = rv[tid + s]; int oi = ri[tid + s];
        if (ov > rv[tid] || (ov == rv[tid] && oi < ri[tid])) { rv[tid] = ov; ri[tid] = oi; }
      }
      __syncthreads();
    }
    if (tid == 0) { tidxv[n * TOPK_ + it] = ri[0]; sv[ri[0]] = -INFINITY; }
    __syncthreads();
  }
}

// ---- gather top-k columns of x and pos_embeddings ----
__global__ void k_gather(const float* __restrict__ x, const float* __restrict__ pos,
                         const int* __restrict__ tidxv,
                         float* __restrict__ oprop, float* __restrict__ opos) {
  int t = blockIdx.x * blockDim.x + threadIdx.x;
  if (t >= B_ * C_ * TOPK_) return;
  int j = t % TOPK_;
  int c = (t / TOPK_) % C_;
  int n = t / (TOPK_ * C_);
  int idx = tidxv[n * TOPK_ + j];
  oprop[t] = x[((size_t)n * C_ + c) * HW_ + idx];
  opos[t]  = pos[(size_t)c * HW_ + idx];
}

extern "C" void kernel_launch(void* const* d_in, const int* in_sizes, int n_in,
                              void* d_out, int out_size, void* d_ws, size_t ws_size,
                              hipStream_t stream) {
  (void)in_sizes; (void)n_in; (void)out_size; (void)ws_size;
  const float* x   = (const float*)d_in[0];
  const float* pos = (const float*)d_in[1];
  const float* w1  = (const float*)d_in[2];
  const float* b1  = (const float*)d_in[3];
  const float* w2  = (const float*)d_in[4];
  const float* b2  = (const float*)d_in[5];

  char* ws = (char*)d_ws;
  unsigned short* xb   = (unsigned short*)(ws + OFF_XB);
  unsigned short* hidb = (unsigned short*)(ws + OFF_HIDB);
  unsigned short* wb1  = (unsigned short*)(ws + OFF_WB1);
  unsigned short* wb2  = (unsigned short*)(ws + OFF_WB2);
  float* pmap = (float*)(ws + OFF_P);
  int*   imap = (int*)  (ws + OFF_I);
  float* smap = (float*)(ws + OFF_S);
  int*   tidx = (int*)  (ws + OFF_T);

  float* out    = (float*)d_out;
  float* oprop  = out;                 // (8,256,100)
  float* opos   = out + 204800;        // (8,256,100)
  float* logits = out + 409600;        // (8,81,100,152)

  k_xpose <<<dim3(10, 16, B_ * H_), dim3(16, 16), 0, stream>>>(x, xb);
  k_wpack <<<2304, 256, 0, stream>>>(w1, w2, wb1, wb2);
  k_conv3 <<<dim3(3, H_, B_), 256, 0, stream>>>(xb, wb1, b1, hidb);
  k_conv2 <<<NPIX / 128, 256, 0, stream>>>(hidb, wb2, b2, logits);
  k_probs <<<(NPIX + 255) / 256, 256, 0, stream>>>(logits, pmap, imap);
  k_score <<<(NPIX + 255) / 256, 256, 0, stream>>>(pmap, imap, smap);
  k_topk  <<<B_, 1024, 0, stream>>>(smap, tidx);
  k_gather<<<(B_ * C_ * TOPK_ + 255) / 256, 256, 0, stream>>>(x, pos, tidx, oprop, opos);
}
